// Net_59090160058989
// MI455X (gfx1250) — matrix operand order repaired and checked by reading, not compiled, not run on hardware
//
#include <hip/hip_runtime.h>
#include <hip/hip_bf16.h>
#include <math.h>

// ---------------------------------------------------------------------------
// ParticleNet-style pipeline for MI455X (gfx1250, wave32, WMMA).
//   K=24 neighbors, H=128, G=48 graphs, 192 lc pts/graph, 96 ts pts/graph.
// EdgeConv refactored as  elu(u_i + v_j)  with u = x_tgt@(Wa-Wb)+b, v = x_src@Wb,
// so all heavy math is N x 128 @ 128 x 128 GEMMs -> v_wmma_f32_16x16x32_f16.
// B kept TRANSPOSED in LDS so fragments load as contiguous ds_load_b128 pairs.
// ---------------------------------------------------------------------------

typedef __attribute__((ext_vector_type(16))) _Float16 v16h;
typedef __attribute__((ext_vector_type(8)))  _Float16 v8h;
typedef __attribute__((ext_vector_type(8)))  float    v8f;

#define GG    48
#define HH    128
#define KNN_K 24
#define NTS   (48 * 96)    // 4608
#define NLC   (48 * 192)   // 9216

__device__ __forceinline__ float elu_f(float x) {
    return x > 0.0f ? x : expm1f(x);
}

// ---------------------------------------------------------------------------
// Layer-1 encoder: out[M][128] = elu(x[M][Kin] @ W[Kin][128] + b), Kin in {6,8}
// ---------------------------------------------------------------------------
__global__ __launch_bounds__(256) void lin_in_kernel(
    const float* __restrict__ x, const float* __restrict__ W,
    const float* __restrict__ b, float* __restrict__ out, int M, int Kin)
{
    int row = blockIdx.x * 2 + (threadIdx.x >> 7);
    int c   = threadIdx.x & 127;
    if (row >= M) return;
    float acc = b[c];
    for (int k = 0; k < Kin; ++k)
        acc = fmaf(x[(size_t)row * Kin + k], W[k * HH + c], acc);
    out[(size_t)row * HH + c] = elu_f(acc);
}

// ---------------------------------------------------------------------------
// Wd = conv_W[0:128,:] - conv_W[128:256,:]
// ---------------------------------------------------------------------------
__global__ void wdiff_kernel(const float* __restrict__ W, float* __restrict__ Wd)
{
    int i = blockIdx.x * 256 + threadIdx.x;
    if (i < HH * HH) Wd[i] = W[i] - W[HH * HH + i];
}

// ---------------------------------------------------------------------------
// WMMA GEMM: C[M][128] = act(A[M][128] @ B[128][128] + bias)
// block = 256 thr (8 waves), computes 64 rows x 128 cols.
// As[row][k] row-major; BsT[n][k] TRANSPOSED -> all fragment loads contiguous.
// Row stride 136 halves (=272B): 16B-aligned rows, near-conflict-free banks.
// wave w -> row-tile (w&3), col-tiles ((w>>2)*4 ..+3); K = 4 chunks of 32.
// ---------------------------------------------------------------------------
__global__ __launch_bounds__(256) void gemm128_kernel(
    const float* __restrict__ A, const float* __restrict__ B,
    const float* __restrict__ bias, float* __restrict__ C, int M, int act)
{
    __shared__ alignas(16) _Float16 As[64][136];
    __shared__ alignas(16) _Float16 BsT[128][136];   // BsT[n][k] = B[k][n]

    const int tid  = threadIdx.x;
    const int row0 = blockIdx.x * 64;

    for (int i = tid; i < HH * HH; i += 256) {
        int k = i >> 7, n = i & 127;
        BsT[n][k] = (_Float16)B[i];                  // coalesced global read
    }
    for (int i = tid; i < 64 * HH; i += 256) {
        int r = i >> 7, k = i & 127;
        As[r][k] = (_Float16)A[(size_t)(row0 + r) * HH + k];
    }
    __syncthreads();

    const int wave = tid >> 5, lane = tid & 31;
    const int rt   = wave & 3;              // row tile 0..3
    const int ch   = wave >> 2;             // col half 0..1
    const int m    = lane & 15;
    const int koff = (lane < 16) ? 0 : 8;   // A-fragment K offset (ISA layout)
    const int kgB  = (lane >> 4) * 8;       // B-fragment K offset (ISA layout)
    const int arow = rt * 16 + m;

    v8f acc[4] = {v8f{}, v8f{}, v8f{}, v8f{}};

    #pragma unroll
    for (int kc = 0; kc < 4; ++kc) {
        const int kb = kc * 32;
        // A fragment: two contiguous 16B LDS loads
        v8h alo = *(const v8h*)&As[arow][kb + koff];
        v8h ahi = *(const v8h*)&As[arow][kb + 16 + koff];
        v16h a  = __builtin_shufflevector(alo, ahi,
                    0,1,2,3,4,5,6,7,8,9,10,11,12,13,14,15);
        #pragma unroll
        for (int nt = 0; nt < 4; ++nt) {
            const int col = (ch * 4 + nt) * 16 + m;
            // B fragment: same K layout as A = two 16B LDS loads
            v8h blo = *(const v8h*)&BsT[col][kb + kgB];
            v8h bhi = *(const v8h*)&BsT[col][kb + 16 + kgB];
            v16h bf = __builtin_shufflevector(blo, bhi,
                        0,1,2,3,4,5,6,7,8,9,10,11,12,13,14,15);
            acc[nt] = __builtin_amdgcn_wmma_f32_16x16x32_f16(
                false, a, false, bf, (short)0, acc[nt], false, false);
        }
    }

    const int mrow = (lane >> 4) * 8;   // C layout: lanes 16-31 hold M+8
    #pragma unroll
    for (int nt = 0; nt < 4; ++nt) {
        const int col = (ch * 4 + nt) * 16 + m;
        const float bv = bias ? bias[col] : 0.0f;
        #pragma unroll
        for (int r8 = 0; r8 < 8; ++r8) {
            const int row = row0 + rt * 16 + mrow + r8;
            float v = acc[nt][r8] + bv;
            if (act) v = elu_f(v);
            C[(size_t)row * HH + col] = v;
        }
    }
}

// ---------------------------------------------------------------------------
// KNN: per graph g, per target t: indices of the 24 smallest ||x_tgt-x_src||^2
// among that graph's n_src sources.  grid = (48, n_tgt/32), block = 256 (8 waves),
// wave handles 4 targets.  Source features staged in LDS (stride 129 floats).
// ---------------------------------------------------------------------------
__global__ __launch_bounds__(256) void knn_kernel(
    const float* __restrict__ xsrc, const float* __restrict__ xtgt,
    int* __restrict__ idx_out, int n_src, int n_tgt)
{
    extern __shared__ float sf[];            // n_src * 129 floats
    const int g   = blockIdx.x;
    const int tid = threadIdx.x;

    for (int i = tid; i < n_src * HH; i += 256) {
        int r = i >> 7, d = i & 127;
        sf[r * 129 + d] = xsrc[((size_t)g * n_src + r) * HH + d];
    }
    __syncthreads();

    const int wave = tid >> 5, lane = tid & 31;
    const int nj   = n_src >> 5;             // 3 (96 srcs) or 6 (192 srcs)
    const int tb   = blockIdx.y * 32;

    for (int tt = wave; tt < 32; tt += 8) {
        const int t = tb + tt;
        const float* trow = xtgt + ((size_t)g * n_tgt + t) * HH;
        float xt[4];
        #pragma unroll
        for (int q = 0; q < 4; ++q) xt[q] = trow[lane + 32 * q];

        float acc[6] = {0.f, 0.f, 0.f, 0.f, 0.f, 0.f};
        #pragma unroll
        for (int q = 0; q < 4; ++q) {
            for (int dd = 0; dd < 32; ++dd) {
                const float xtd = __shfl(xt[q], dd);
                const int   d   = q * 32 + dd;
                #pragma unroll
                for (int jj = 0; jj < 6; ++jj)
                    if (jj < nj) {
                        float diff = sf[(lane + (jj << 5)) * 129 + d] - xtd;
                        acc[jj] = fmaf(diff, diff, acc[jj]);
                    }
            }
        }

        const int tg = g * n_tgt + t;
        for (int k = 0; k < KNN_K; ++k) {
            float bv = __builtin_inff();
            int   bj = 0x7fffffff;
            #pragma unroll
            for (int jj = 0; jj < 6; ++jj)
                if (jj < nj) {
                    int j = lane + (jj << 5);
                    if (acc[jj] < bv || (acc[jj] == bv && j < bj)) { bv = acc[jj]; bj = j; }
                }
            #pragma unroll
            for (int off = 16; off > 0; off >>= 1) {
                float ov = __shfl_xor(bv, off);
                int   oj = __shfl_xor(bj, off);
                if (ov < bv || (ov == bv && oj < bj)) { bv = ov; bj = oj; }
            }
            const int jd = bj >> 5;
            #pragma unroll
            for (int jj = 0; jj < 6; ++jj)
                if (jj == jd && (bj & 31) == lane) acc[jj] = __builtin_inff();
            if (lane == 0) idx_out[tg * KNN_K + k] = g * n_src + bj;
        }
    }
}

// ---------------------------------------------------------------------------
// out[t][c] = max_k elu(u[t][c] + v[idx[t][k]][c])   (block = target, 128 thr)
// ---------------------------------------------------------------------------
__global__ __launch_bounds__(128) void edge_agg_kernel(
    const float* __restrict__ u, const float* __restrict__ v,
    const int* __restrict__ idx, float* __restrict__ out)
{
    const int t = blockIdx.x, c = threadIdx.x;
    const float uval = u[(size_t)t * HH + c];
    float m = -__builtin_inff();
    for (int k = 0; k < KNN_K; ++k) {
        const int j = idx[t * KNN_K + k];
        m = fmaxf(m, elu_f(uval + v[(size_t)j * HH + c]));
    }
    out[(size_t)t * HH + c] = m;
}

// ---------------------------------------------------------------------------
// Mean-pool the 192 lc rows of each graph.
// ---------------------------------------------------------------------------
__global__ __launch_bounds__(128) void pool_kernel(
    const float* __restrict__ f3, float* __restrict__ pooled)
{
    const int g = blockIdx.x, c = threadIdx.x;
    float s = 0.0f;
    for (int i = 0; i < 192; ++i)
        s += f3[((size_t)g * 192 + i) * HH + c];
    pooled[g * HH + c] = s * (1.0f / 192.0f);
}

// ---------------------------------------------------------------------------
// Head MLP 128->64->32->8->5 (+ arange output). One block per graph.
// ---------------------------------------------------------------------------
__global__ __launch_bounds__(128) void head_kernel(
    const float* __restrict__ pooled,
    const float* __restrict__ W1, const float* __restrict__ b1,
    const float* __restrict__ W2, const float* __restrict__ b2,
    const float* __restrict__ W3, const float* __restrict__ b3,
    const float* __restrict__ W4, const float* __restrict__ b4,
    float* __restrict__ out)
{
    __shared__ float h0[128], h1[64], h2[32], h3[8];
    const int g = blockIdx.x, tid = threadIdx.x;
    h0[tid] = pooled[g * HH + tid];
    __syncthreads();
    if (tid < 64) {
        float a = b1[tid];
        for (int k = 0; k < 128; ++k) a = fmaf(h0[k], W1[k * 64 + tid], a);
        h1[tid] = elu_f(a);
    }
    __syncthreads();
    if (tid < 32) {
        float a = b2[tid];
        for (int k = 0; k < 64; ++k) a = fmaf(h1[k], W2[k * 32 + tid], a);
        h2[tid] = elu_f(a);
    }
    __syncthreads();
    if (tid < 8) {
        float a = b3[tid];
        for (int k = 0; k < 32; ++k) a = fmaf(h2[k], W3[k * 8 + tid], a);
        h3[tid] = elu_f(a);
    }
    __syncthreads();
    if (tid < 5) {
        float a = b4[tid];
        for (int k = 0; k < 8; ++k) a = fmaf(h3[k], W4[k * 5 + tid], a);
        out[g * 5 + tid] = a;
    }
    if (tid == 0) out[GG * 5 + g] = (float)g;   // second tuple output: arange(G)
}

// ---------------------------------------------------------------------------
// Host orchestration
// ---------------------------------------------------------------------------
extern "C" void kernel_launch(void* const* d_in, const int* in_sizes, int n_in,
                              void* d_out, int out_size, void* d_ws, size_t ws_size,
                              hipStream_t stream)
{
    (void)in_sizes; (void)n_in; (void)out_size; (void)ws_size;

    const float* x_ts   = (const float*)d_in[0];
    const float* x_lc   = (const float*)d_in[1];
    // d_in[2] x_ass, d_in[3..5] batch arrays: unused (regular batch structure)
    const float* ts_W1  = (const float*)d_in[6];
    const float* ts_b1  = (const float*)d_in[7];
    const float* ts_W2  = (const float*)d_in[8];
    const float* ts_b2  = (const float*)d_in[9];
    const float* lc_W1  = (const float*)d_in[10];
    const float* lc_b1  = (const float*)d_in[11];
    const float* lc_W2  = (const float*)d_in[12];
    const float* lc_b2  = (const float*)d_in[13];
    const float* conv_W = (const float*)d_in[14];
    const float* conv_b = (const float*)d_in[15];
    const float* o_W1   = (const float*)d_in[16];
    const float* o_b1   = (const float*)d_in[17];
    const float* o_W2   = (const float*)d_in[18];
    const float* o_b2   = (const float*)d_in[19];
    const float* o_W3   = (const float*)d_in[20];
    const float* o_b3   = (const float*)d_in[21];
    const float* o_W4   = (const float*)d_in[22];
    const float* o_b4   = (const float*)d_in[23];
    float* out = (float*)d_out;

    // --- workspace carve-up (floats) ---
    float* p = (float*)d_ws;
    float* Wd     = p;  p += HH * HH;
    float* hid    = p;  p += (size_t)NLC * HH;     // layer-1 scratch (max size)
    float* ts_enc = p;  p += (size_t)NTS * HH;
    float* lc_enc = p;  p += (size_t)NLC * HH;
    float* f1     = p;  p += (size_t)NLC * HH;
    float* f11    = p;  p += (size_t)NLC * HH;
    float* f2     = p;  p += (size_t)NTS * HH;
    float* f22    = p;  p += (size_t)NTS * HH;
    float* f3     = p;  p += (size_t)NLC * HH;
    float* ubuf   = p;  p += (size_t)NLC * HH;
    float* vbuf   = p;  p += (size_t)NLC * HH;
    float* pooled = p;  p += GG * HH;
    int*   idx    = (int*)p;                       // NLC * 24 ints

    const float* Wb = conv_W + HH * HH;            // rows 128..255 of conv_W

    // Wd = Wa - Wb
    wdiff_kernel<<<(HH * HH + 255) / 256, 256, 0, stream>>>(conv_W, Wd);

    // encoders
    lin_in_kernel<<<NTS / 2, 256, 0, stream>>>(x_ts, ts_W1, ts_b1, hid, NTS, 6);
    gemm128_kernel<<<NTS / 64, 256, 0, stream>>>(hid, ts_W2, ts_b2, ts_enc, NTS, 1);
    lin_in_kernel<<<NLC / 2, 256, 0, stream>>>(x_lc, lc_W1, lc_b1, hid, NLC, 8);
    gemm128_kernel<<<NLC / 64, 256, 0, stream>>>(hid, lc_W2, lc_b2, lc_enc, NLC, 1);

    // EdgeConv: knn -> u = tgt@Wd + b, v = src@Wb -> max_k elu(u+v)
    auto conv = [&](const float* src, const float* tgt, int nsp, int ntp, float* outf) {
        const int nsrc_tot = GG * nsp, ntgt_tot = GG * ntp;
        const size_t shm = (size_t)nsp * 129 * sizeof(float);
        knn_kernel<<<dim3(GG, ntp / 32), 256, shm, stream>>>(src, tgt, idx, nsp, ntp);
        gemm128_kernel<<<ntgt_tot / 64, 256, 0, stream>>>(tgt, Wd, conv_b, ubuf, ntgt_tot, 0);
        gemm128_kernel<<<nsrc_tot / 64, 256, 0, stream>>>(src, Wb, nullptr, vbuf, nsrc_tot, 0);
        edge_agg_kernel<<<ntgt_tot, 128, 0, stream>>>(ubuf, vbuf, idx, outf);
    };

    conv(lc_enc, lc_enc, 192, 192, f1);    // f1
    conv(f1,     f1,     192, 192, f11);   // f11
    conv(ts_enc, ts_enc,  96,  96, f2);    // f2
    conv(f2,     f2,      96,  96, f22);   // f22
    conv(f22,    f11,     96, 192, f3);    // f3 (ts sources -> lc targets)

    pool_kernel<<<GG, 128, 0, stream>>>(f3, pooled);
    head_kernel<<<GG, 128, 0, stream>>>(pooled, o_W1, o_b1, o_W2, o_b2,
                                        o_W3, o_b3, o_W4, o_b4, out);
}
